// RNNLayer_40329742910037
// MI455X (gfx1250) — matrix operand order repaired and checked by reading, not compiled, not run on hardware
//
#include <hip/hip_runtime.h>

typedef __attribute__((ext_vector_type(16))) _Float16 v16h;
typedef __attribute__((ext_vector_type(8)))  _Float16 v8h;
typedef __attribute__((ext_vector_type(8)))  float    v8f;

constexpr int B_ = 64, T_ = 512, I_ = 512, H_ = 512;
constexpr int G4H = 4 * H_;            // 2048
constexpr long ROWS = (long)B_ * T_;   // 32768

struct Frag { v8h h[2]; };             // one 16-lane-striped WMMA operand (8 VGPRs)

// ---------- helpers ----------
__device__ __forceinline__ unsigned encf(float f) {
    unsigned u = __float_as_uint(f);
    return (u & 0x80000000u) ? ~u : (u | 0x80000000u);
}
__device__ __forceinline__ float decf(unsigned e) {
    unsigned u = (e & 0x80000000u) ? (e ^ 0x80000000u) : ~e;
    return __uint_as_float(u);
}
__device__ __forceinline__ float fq(float x, float s, float z) {
    float q = rintf(x / s) + z;                 // round half-to-even, matches jnp.round
    q = fminf(fmaxf(q, -128.0f), 127.0f);
    return (q - z) * s;
}
__device__ __forceinline__ float sigm(float x) { return 1.0f / (1.0f + expf(-x)); }

__device__ __forceinline__ v16h joinFrag(const Frag& f) {
    union { v16h v; v8h h[2]; } u;
    u.h[0] = f.h[0]; u.h[1] = f.h[1];
    return u.v;
}

// ---------- min/max reduction ----------
__global__ void init_hdr_kernel(unsigned* mm) {
    int t = threadIdx.x;
    if (t < 4) { mm[2 * t] = 0xFFFFFFFFu; mm[2 * t + 1] = 0u; }
}

__global__ void minmax_kernel(const float* __restrict__ x, long n, unsigned* mm) {
    __shared__ unsigned smin[256], smax[256];
    unsigned lmin = 0xFFFFFFFFu, lmax = 0u;
    for (long i = (long)blockIdx.x * blockDim.x + threadIdx.x; i < n;
         i += (long)gridDim.x * blockDim.x) {
        unsigned e = encf(x[i]);
        lmin = min(lmin, e); lmax = max(lmax, e);
    }
    smin[threadIdx.x] = lmin; smax[threadIdx.x] = lmax;
    __syncthreads();
    for (int s = 128; s > 0; s >>= 1) {
        if (threadIdx.x < (unsigned)s) {
            smin[threadIdx.x] = min(smin[threadIdx.x], smin[threadIdx.x + s]);
            smax[threadIdx.x] = max(smax[threadIdx.x], smax[threadIdx.x + s]);
        }
        __syncthreads();
    }
    if (threadIdx.x == 0) { atomicMin(&mm[0], smin[0]); atomicMax(&mm[1], smax[0]); }
}

__global__ void qparams_kernel(const unsigned* __restrict__ mm, float* __restrict__ qp,
                               int slot0, int nslots) {
    int t = threadIdx.x;
    if (t < nslots) {
        int s = slot0 + t;
        float mn = decf(mm[2 * s]);
        float mx = decf(mm[2 * s + 1]);
        float scale = fmaxf((mx - mn) / 255.0f, 1e-8f);
        float zp = fminf(fmaxf(rintf(-128.0f - mn / scale), -128.0f), 127.0f);
        qp[2 * s] = scale; qp[2 * s + 1] = zp;
    }
}

// ---------- conversions ----------
__global__ void f32_to_f16_kernel(const float* __restrict__ src, _Float16* __restrict__ dst, long n) {
    for (long i = (long)blockIdx.x * blockDim.x + threadIdx.x; i < n;
         i += (long)gridDim.x * blockDim.x)
        dst[i] = (_Float16)src[i];
}

__global__ void fq_to_f16_kernel(const float* __restrict__ src, _Float16* __restrict__ dst,
                                 long n, const float* __restrict__ qp) {
    float s = qp[0], z = qp[1];
    for (long i = (long)blockIdx.x * blockDim.x + threadIdx.x; i < n;
         i += (long)gridDim.x * blockDim.x)
        dst[i] = (_Float16)fq(src[i], s, z);
}

__global__ void fq_inplace_kernel(float* __restrict__ x, long n, const float* __restrict__ qp) {
    float s = qp[0], z = qp[1];
    for (long i = (long)blockIdx.x * blockDim.x + threadIdx.x; i < n;
         i += (long)gridDim.x * blockDim.x)
        x[i] = fq(x[i], s, z);
}

// ---------- x_gates = inputs_q @ W_ih^T + b_ih  (32768x512 * 512x2048) ----------
// One wave computes a 16x64 strip (4 N-tiles); A + 4xB fragments are two-stage
// software pipelined across the K loop so loads overlap the 4 WMMAs.
__global__ void xgates_wmma_kernel(const _Float16* __restrict__ Aq,   // [ROWS][I]
                                   const _Float16* __restrict__ Wih,  // [4H][I]
                                   const float*    __restrict__ bih,  // [4H]
                                   _Float16*       __restrict__ XG) { // [ROWS][4H]
    int lane = threadIdx.x & 31;
    long gwave = (long)blockIdx.x * (blockDim.x >> 5) + (threadIdx.x >> 5);
    int rowTile = (int)(gwave >> 5);   // 0..2047
    int nGroup  = (int)(gwave & 31);   // 0..31
    int r0 = rowTile * 16;
    int n0 = nGroup * 64;
    int m  = lane & 15;
    int kh = lane >> 4;

    const _Float16* aRow = Aq + (long)(r0 + m) * I_ + kh * 8;        // + k0
    const _Float16* bRow[4];
#pragma unroll
    for (int j = 0; j < 4; ++j)
        bRow[j] = Wih + (long)(n0 + j * 16 + m) * I_ + kh * 8;       // + k0

    Frag aC, aN, bC[4], bN[4];
    // prologue: k0 = 0
    aC.h[0] = *(const v8h*)(aRow);
    aC.h[1] = *(const v8h*)(aRow + 16);
#pragma unroll
    for (int j = 0; j < 4; ++j) {
        bC[j].h[0] = *(const v8h*)(bRow[j]);
        bC[j].h[1] = *(const v8h*)(bRow[j] + 16);
    }

    v8f acc[4] = {};
#pragma unroll
    for (int k0 = 0; k0 < I_; k0 += 32) {
        int kn = (k0 + 32 < I_) ? (k0 + 32) : k0;   // harmless reload on last iter
        __builtin_prefetch(aRow + kn + 32, 0, 3);
        aN.h[0] = *(const v8h*)(aRow + kn);
        aN.h[1] = *(const v8h*)(aRow + kn + 16);
#pragma unroll
        for (int j = 0; j < 4; ++j) {
            bN[j].h[0] = *(const v8h*)(bRow[j] + kn);
            bN[j].h[1] = *(const v8h*)(bRow[j] + kn + 16);
        }
#pragma unroll
        for (int j = 0; j < 4; ++j)
            acc[j] = __builtin_amdgcn_wmma_f32_16x16x32_f16(
                false, joinFrag(aC), false, joinFrag(bC[j]), (short)0, acc[j],
                false, false);
        aC = aN;
#pragma unroll
        for (int j = 0; j < 4; ++j) bC[j] = bN[j];
    }

    int rbase = kh * 8;  // D layout: VGPR r -> M = r + 8*(lane>=16), N = lane&15
#pragma unroll
    for (int j = 0; j < 4; ++j) {
        int n = n0 + j * 16 + m;
        float bias = bih[n];
#pragma unroll
        for (int r = 0; r < 8; ++r) {
            long row = r0 + rbase + r;
            XG[row * G4H + n] = (_Float16)(acc[j][r] + bias);
        }
    }
}

// ---------- fused per-timestep kernel ----------
// grid(32 hTiles, 4 bTiles), 128 threads = 4 waves (wave g handles gate g).
__global__ void lstm_step_kernel(const float* __restrict__ h_in,   // [B][H] (read)
                                 float*       __restrict__ h_out,  // [B][H] (write)
                                 float*       __restrict__ c_buf,  // [B][H] (rmw, tile-local)
                                 const _Float16* __restrict__ Whh, // [4H][H]
                                 const float* __restrict__ bhh,    // [4H]
                                 const _Float16* __restrict__ XG,  // [ROWS][4H]
                                 const float* __restrict__ qp,     // qparams header
                                 float*       __restrict__ outp,   // [B][T][H]
                                 int t) {
    __shared__ _Float16 ldsA[16][520];     // fq(h) rows for this bTile (padded stride)
    __shared__ float    ldsG[4][16][16];   // per-gate accum tiles
    int hTile = blockIdx.x;                // 0..31
    int bTile = blockIdx.y;                // 0..3
    int b0 = bTile * 16;
    int tid = threadIdx.x;
    float h_s = qp[2], h_z = qp[3];
    float c_s = qp[4], c_z = qp[5];

    // Phase 1: fake-quant h rows -> LDS (f16); vectorized 128-bit loads
    for (int i4 = tid; i4 < 16 * (H_ / 4); i4 += 128) {   // 2048 float4s
        int row = i4 >> 7;                 // 128 float4 per row
        int col = (i4 & 127) * 4;
        float4 v = *(const float4*)&h_in[(b0 + row) * H_ + col];
        ldsA[row][col + 0] = (_Float16)fq(v.x, h_s, h_z);
        ldsA[row][col + 1] = (_Float16)fq(v.y, h_s, h_z);
        ldsA[row][col + 2] = (_Float16)fq(v.z, h_s, h_z);
        ldsA[row][col + 3] = (_Float16)fq(v.w, h_s, h_z);
    }
    __syncthreads();

    // Phase 2: wave g computes 16x16 tile of gate g via WMMA over K=512.
    // B (W_hh) fragments are two-stage pipelined; A comes from LDS (ds_load_b128).
    int g    = tid >> 5;
    int lane = tid & 31;
    int m    = lane & 15;
    int kh   = lane >> 4;
    const _Float16* bRow = Whh + (long)(g * H_ + hTile * 16 + m) * H_ + kh * 8;

    Frag bC, bN;
    bC.h[0] = *(const v8h*)(bRow);
    bC.h[1] = *(const v8h*)(bRow + 16);

    v8f acc = {};
#pragma unroll
    for (int k0 = 0; k0 < H_; k0 += 32) {
        int kn = (k0 + 32 < H_) ? (k0 + 32) : k0;
        bN.h[0] = *(const v8h*)(bRow + kn);
        bN.h[1] = *(const v8h*)(bRow + kn + 16);
        union { v16h v; v8h h[2]; } a;
        a.h[0] = *(const v8h*)&ldsA[m][k0 + kh * 8];        // ds_load_b128
        a.h[1] = *(const v8h*)&ldsA[m][k0 + 16 + kh * 8];
        acc = __builtin_amdgcn_wmma_f32_16x16x32_f16(
            false, a.v, false, joinFrag(bC), (short)0, acc, false, false);
        bC = bN;
    }
#pragma unroll
    for (int r = 0; r < 8; ++r)
        ldsG[g][kh * 8 + r][m] = acc[r];
    __syncthreads();

    // Phase 3: LSTM elementwise (256 outputs, 2 per thread)
    for (int i = tid; i < 256; i += 128) {
        int row = i >> 4, col = i & 15;
        int b  = b0 + row;
        int hh = hTile * 16 + col;
        long xbase = ((long)b * T_ + t) * G4H;
        float iv = ldsG[0][row][col] + (float)XG[xbase + 0 * H_ + hh] + bhh[0 * H_ + hh];
        float fv = ldsG[1][row][col] + (float)XG[xbase + 1 * H_ + hh] + bhh[1 * H_ + hh];
        float gv = ldsG[2][row][col] + (float)XG[xbase + 2 * H_ + hh] + bhh[2 * H_ + hh];
        float ov = ldsG[3][row][col] + (float)XG[xbase + 3 * H_ + hh] + bhh[3 * H_ + hh];
        float cq = fq(c_buf[b * H_ + hh], c_s, c_z);
        float cn = sigm(fv) * cq + sigm(iv) * tanhf(gv);
        float hn = sigm(ov) * tanhf(cn);
        c_buf[b * H_ + hh] = cn;
        h_out[b * H_ + hh] = hn;
        outp[((long)b * T_ + t) * H_ + hh] = hn;
    }
}

// ---------- host ----------
extern "C" void kernel_launch(void* const* d_in, const int* in_sizes, int n_in,
                              void* d_out, int out_size, void* d_ws, size_t ws_size,
                              hipStream_t stream) {
    (void)in_sizes; (void)n_in; (void)out_size; (void)ws_size;
    const float* inputs = (const float*)d_in[0];
    const float* h_prev = (const float*)d_in[1];
    const float* c_prev = (const float*)d_in[2];
    const float* W_ih   = (const float*)d_in[3];
    const float* W_hh   = (const float*)d_in[4];
    const float* b_ih   = (const float*)d_in[5];
    const float* b_hh   = (const float*)d_in[6];

    float* out     = (float*)d_out;
    float* outputs = out;                                 // [B][T][H]
    float* h_last  = out + (long)B_ * T_ * H_;
    float* c_last  = h_last + (long)B_ * H_;

    char* ws = (char*)d_ws;
    unsigned* mm = (unsigned*)ws;                         // 8 x u32 : {min,max} for in/h/c/out
    float*    qp = (float*)(ws + 32);                     // 8 x f32 : {scale,zp} for in/h/c/out
    size_t off = 256;
    _Float16* inF  = (_Float16*)(ws + off); off += (size_t)ROWS * I_ * 2;
    _Float16* WihF = (_Float16*)(ws + off); off += (size_t)G4H * I_ * 2;
    _Float16* WhhF = (_Float16*)(ws + off); off += (size_t)G4H * H_ * 2;
    _Float16* XG   = (_Float16*)(ws + off); off += (size_t)ROWS * G4H * 2;
    float* hbuf0 = (float*)(ws + off); off += (size_t)B_ * H_ * 4;
    float* hbuf1 = (float*)(ws + off); off += (size_t)B_ * H_ * 4;
    float* cbuf  = (float*)(ws + off); off += (size_t)B_ * H_ * 4;

    // 1) observers for inputs / h_prev / c_prev
    init_hdr_kernel<<<1, 32, 0, stream>>>(mm);
    minmax_kernel<<<1024, 256, 0, stream>>>(inputs, ROWS * (long)I_, mm + 0);
    minmax_kernel<<<64, 256, 0, stream>>>(h_prev, (long)B_ * H_, mm + 2);
    minmax_kernel<<<64, 256, 0, stream>>>(c_prev, (long)B_ * H_, mm + 4);
    qparams_kernel<<<1, 32, 0, stream>>>(mm, qp, 0, 3);

    // 2) weight conversion + fake-quant inputs to f16
    f32_to_f16_kernel<<<512, 256, 0, stream>>>(W_ih, WihF, (long)G4H * I_);
    f32_to_f16_kernel<<<512, 256, 0, stream>>>(W_hh, WhhF, (long)G4H * H_);
    fq_to_f16_kernel<<<2048, 256, 0, stream>>>(inputs, inF, ROWS * (long)I_, qp + 0);

    hipMemcpyAsync(hbuf0, h_prev, (size_t)B_ * H_ * 4, hipMemcpyDeviceToDevice, stream);
    hipMemcpyAsync(cbuf,  c_prev, (size_t)B_ * H_ * 4, hipMemcpyDeviceToDevice, stream);

    // 3) big parallel GEMM: x_gates (65536 waves, 8 waves/block)
    xgates_wmma_kernel<<<8192, 256, 0, stream>>>(inF, WihF, b_ih, XG);

    // 4) sequential scan; double-buffer h across steps
    dim3 sgrid(32, 4);
    for (int t = 0; t < T_; ++t) {
        const float* hin = (t & 1) ? hbuf1 : hbuf0;
        float*      hout = (t & 1) ? hbuf0 : hbuf1;
        lstm_step_kernel<<<sgrid, 128, 0, stream>>>(hin, hout, cbuf, WhhF, b_hh,
                                                    XG, qp, outputs, t);
    }
    // t=511 wrote hbuf0
    hipMemcpyAsync(h_last, hbuf0, (size_t)B_ * H_ * 4, hipMemcpyDeviceToDevice, stream);
    hipMemcpyAsync(c_last, cbuf,  (size_t)B_ * H_ * 4, hipMemcpyDeviceToDevice, stream);

    // 5) output observer + fake-quant outputs in place
    minmax_kernel<<<1024, 256, 0, stream>>>(outputs, (long)B_ * T_ * H_, mm + 6);
    qparams_kernel<<<1, 32, 0, stream>>>(mm, qp, 3, 1);
    fq_inplace_kernel<<<2048, 256, 0, stream>>>(outputs, (long)B_ * T_ * H_, qp + 6);
}
